// LinearAttention_76381698392312
// MI455X (gfx1250) — matrix operand order repaired and checked by reading, not compiled, not run on hardware
//
#include <hip/hip_runtime.h>

// ---------------------------------------------------------------------------
// Linear attention for MI455X (gfx1250, wave32, WMMA).
// All GEMMs use v_wmma_f32_16x16x32_f16; all LDS fragment loads are b128.
// ---------------------------------------------------------------------------

typedef __attribute__((ext_vector_type(16))) _Float16 v16h;
typedef __attribute__((ext_vector_type(8)))  float    v8f;
typedef __attribute__((ext_vector_type(8)))  _Float16 h8;

#define BQ 4
#define TQ 4096
#define DQ 1024
#define HQ 16
#define DKQ 64
#define EPSQ 1e-6f

__device__ __forceinline__ v8f wmma_f16(v16h a, v16h b, v8f c) {
  // (neg_a, A, neg_b, B, c_mod, C, reuse_a, reuse_b)
  return __builtin_amdgcn_wmma_f32_16x16x32_f16(false, a, false, b, (short)0, c,
                                                false, false);
}

// A fragment: 16x32 (MxK) f16 from LDS tile [row][k] (stride multiple of 8).
// Lane L: m = L%16, half = L/16. VGPR 0-3 = k in [8*half, 8*half+8),
// VGPR 4-7 = k in [16+8*half, 16+8*half+8)  -> two contiguous b128 loads.
__device__ __forceinline__ v16h load_a_frag(const _Float16* lds, int row0,
                                            int koff, int stride) {
  int lane = threadIdx.x & 31;
  const _Float16* base =
      lds + (row0 + (lane & 15)) * stride + koff + 8 * (lane >> 4);
  union { v16h v; h8 q[2]; } u;
  u.q[0] = *(const h8*)(base);
  u.q[1] = *(const h8*)(base + 16);
  return u.v;
}

// B fragment: 32x16 (KxN) f16 from LDS tile staged COLUMN-major [n][k]
// (stride multiple of 8). Lane L: n = col0 + L%16, k = 8*half + [0..8) and
// 16 + 8*half + [0..8) -> two b128 loads (same k order as A).
__device__ __forceinline__ v16h load_b_frag(const _Float16* lds, int col0,
                                            int koff, int stride) {
  int lane = threadIdx.x & 31;
  const _Float16* base =
      lds + (col0 + (lane & 15)) * stride + koff + 8 * (lane >> 4);
  union { v16h v; h8 q[2]; } u;
  u.q[0] = *(const h8*)(base);
  u.q[1] = *(const h8*)(base + 16);
  return u.v;
}

// ---------------------------------------------------------------------------
// f32 -> f16 elementwise convert (x and the four weight matrices)
// ---------------------------------------------------------------------------
__global__ void f32_to_f16_kernel(const float* __restrict__ src,
                                  _Float16* __restrict__ dst, int n) {
  int i = (blockIdx.x * 256 + threadIdx.x) * 4;
  if (i + 3 < n) {
    float4 v = *(const float4*)(src + i);
    dst[i + 0] = (_Float16)v.x;
    dst[i + 1] = (_Float16)v.y;
    dst[i + 2] = (_Float16)v.z;
    dst[i + 3] = (_Float16)v.w;
  }
}

// ---------------------------------------------------------------------------
// Projection GEMM: Y[M=16384, N=1024] = A[M,1024] @ W[N,1024]^T + bias
// W stays in its native [n][k] layout (f16), so B tiles stage with b128.
// Block tile 256x128, 8 waves, wave tile 64x64, K step 32.
// mode 0: f16 headed [B,H,T,DK] out; mode 1: same + phi; mode 2: f32 flat out.
// ---------------------------------------------------------------------------
__global__ __launch_bounds__(256)
void proj_gemm_kernel(const _Float16* __restrict__ A,
                      const _Float16* __restrict__ Wh,
                      const float* __restrict__ bias,
                      _Float16* __restrict__ outH,
                      float* __restrict__ outF, int mode) {
  __shared__ __align__(16) _Float16 aT[256 * 40];  // [row][k], pad 32->40
  __shared__ __align__(16) _Float16 bT[128 * 40];  // [n][k],  pad 32->40

  const int blockM = blockIdx.y * 256;
  const int blockN = blockIdx.x * 128;
  const int tid  = threadIdx.x;
  const int lane = tid & 31;
  const int wave = tid >> 5;
  const int wm = (wave & 3) * 64;  // 4 wave rows of 64
  const int wn = (wave >> 2) * 64; // 2 wave cols of 64

  v8f acc[4][4] = {};

  for (int kk = 0; kk < DQ; kk += 32) {
    __syncthreads();
    // Stage A: 256 rows x 32 k, b128 both sides (4 per thread)
#pragma unroll
    for (int r = 0; r < 4; ++r) {
      int idx = tid + 256 * r;          // 0..1023
      int row = idx >> 2;
      int kc = (idx & 3) * 8;
      *(h8*)(aT + row * 40 + kc) =
          *(const h8*)(A + (size_t)(blockM + row) * DQ + kk + kc);
    }
    // Stage B: 128 n-rows x 32 k, direct from W[n][k] (2 per thread)
#pragma unroll
    for (int r = 0; r < 2; ++r) {
      int idx = tid + 256 * r;          // 0..511
      int n = idx >> 2;
      int kc = (idx & 3) * 8;
      *(h8*)(bT + n * 40 + kc) =
          *(const h8*)(Wh + (size_t)(blockN + n) * DQ + kk + kc);
    }
    if (kk + 32 < DQ) {  // prefetch next K slice (global_prefetch_b8)
      __builtin_prefetch(A + (size_t)(blockM + tid) * DQ + kk + 32, 0, 1);
      __builtin_prefetch(Wh + (size_t)(blockN + (tid & 127)) * DQ + kk + 32, 0, 1);
    }
    __syncthreads();

    v16h a0 = load_a_frag(aT, wm +  0, 0, 40);
    v16h a1 = load_a_frag(aT, wm + 16, 0, 40);
    v16h a2 = load_a_frag(aT, wm + 32, 0, 40);
    v16h a3 = load_a_frag(aT, wm + 48, 0, 40);
#pragma unroll
    for (int j = 0; j < 4; ++j) {
      v16h b = load_b_frag(bT, wn + 16 * j, 0, 40);
      acc[0][j] = wmma_f16(a0, b, acc[0][j]);
      acc[1][j] = wmma_f16(a1, b, acc[1][j]);
      acc[2][j] = wmma_f16(a2, b, acc[2][j]);
      acc[3][j] = wmma_f16(a3, b, acc[3][j]);
    }
  }

  // Epilogue: bias (+ phi). C layout: lane n = L%16; VGPR r: m = r + 8*(L/16).
  const int half = lane >> 4;
  const int nl = lane & 15;
#pragma unroll
  for (int j = 0; j < 4; ++j) {
    int col = blockN + wn + 16 * j + nl;
    float bvv = bias[col];
#pragma unroll
    for (int i = 0; i < 4; ++i) {
#pragma unroll
      for (int r = 0; r < 8; ++r) {
        int row = blockM + wm + 16 * i + r + 8 * half;  // global bt index
        float v = acc[i][j][r] + bvv;
        if (mode == 1) v = (v > 0.f) ? (v + 1.f) : __expf(v);  // elu(v)+1
        if (mode == 2) {
          outF[(size_t)row * DQ + col] = v;
        } else {
          int bb = row >> 12, t = row & (TQ - 1);
          int hh = col >> 6, dk = col & (DKQ - 1);
          outH[(((size_t)bb * HQ + hh) * TQ + t) * DKQ + dk] = (_Float16)v;
        }
      }
    }
  }
}

// ---------------------------------------------------------------------------
// ksum[b,h,d] = sum_t k[b,h,t,d].  One block per (b,h).
// ---------------------------------------------------------------------------
__global__ void ksum_kernel(const _Float16* __restrict__ kh,
                            float* __restrict__ ksum) {
  __shared__ float red[256];
  int bh = blockIdx.x;
  int d = threadIdx.x & 63;
  int c = threadIdx.x >> 6;
  const _Float16* base = kh + (size_t)bh * TQ * DKQ;
  float s = 0.f;
  for (int t = c * 1024; t < (c + 1) * 1024; ++t)
    s += (float)base[(size_t)t * DKQ + d];
  red[threadIdx.x] = s;
  __syncthreads();
  if (c == 0)
    ksum[bh * DKQ + d] = red[d] + red[64 + d] + red[128 + d] + red[192 + d];
}

// ---------------------------------------------------------------------------
// kvT[b,h,m,d] = sum_t k[t,d] * v[t,m]   (stored TRANSPOSED: [m][d])
// A = k^T staged [d][t]; B = v^T staged [m][t] (both transposed in LDS).
// One block per (b,h); 8 waves each own a 16x32 tile of the 64x64 output.
// ---------------------------------------------------------------------------
__global__ void kv_kernel(const _Float16* __restrict__ kh,
                          const _Float16* __restrict__ vh,
                          _Float16* __restrict__ kvT) {
  __shared__ __align__(16) _Float16 aT[64 * 40];  // [d][t-local]
  __shared__ __align__(16) _Float16 bT[64 * 40];  // [m][t-local]
  int bh = blockIdx.x;
  int tid = threadIdx.x;
  int lane = tid & 31;
  int wave = tid >> 5;
  int dTile = (wave & 3) * 16;   // 4 d-tiles of 16
  int mHalf = (wave >> 2) * 32;  // 2 m-halves of 32

  const _Float16* kbase = kh + (size_t)bh * TQ * DKQ;
  const _Float16* vbase = vh + (size_t)bh * TQ * DKQ;

  v8f acc[2] = {};
  for (int kk = 0; kk < TQ; kk += 32) {
    __syncthreads();
    int tl = tid >> 3;          // 32 t rows
    int dc = (tid & 7) * 8;     // 8 chunks of 8
    h8 k8 = *(const h8*)(kbase + (size_t)(kk + tl) * DKQ + dc);
    h8 v8 = *(const h8*)(vbase + (size_t)(kk + tl) * DKQ + dc);
#pragma unroll
    for (int e = 0; e < 8; ++e) {
      aT[(dc + e) * 40 + tl] = k8[e];
      bT[(dc + e) * 40 + tl] = v8[e];
    }
    __syncthreads();

    v16h af = load_a_frag(aT, dTile, 0, 40);
    v16h b0 = load_b_frag(bT, mHalf, 0, 40);
    v16h b1 = load_b_frag(bT, mHalf + 16, 0, 40);
    acc[0] = wmma_f16(af, b0, acc[0]);
    acc[1] = wmma_f16(af, b1, acc[1]);
  }

  int half = lane >> 4;
  int nl = lane & 15;
#pragma unroll
  for (int j = 0; j < 2; ++j) {
#pragma unroll
    for (int r = 0; r < 8; ++r) {
      int drow = dTile + r + 8 * half;
      int mcol = mHalf + j * 16 + nl;
      // transposed store: [m][d]
      kvT[(size_t)bh * DKQ * DKQ + (size_t)mcol * DKQ + drow] =
          (_Float16)acc[j][r];
    }
  }
}

// ---------------------------------------------------------------------------
// z[b,h,t] = 1 / (q_t . (ksum + eps) + eps)
// ---------------------------------------------------------------------------
__global__ void z_kernel(const _Float16* __restrict__ qh,
                         const float* __restrict__ ksum,
                         float* __restrict__ z) {
  int idx = blockIdx.x * 256 + threadIdx.x;  // 0 .. B*H*T-1
  int bh = idx >> 12;
  const _Float16* qrow = qh + (size_t)idx * DKQ;
  const float* ks = ksum + bh * DKQ;
  float s = 0.f;
#pragma unroll 8
  for (int d = 0; d < DKQ; ++d) s += (float)qrow[d] * (ks[d] + EPSQ);
  z[idx] = 1.0f / (s + EPSQ);
}

// ---------------------------------------------------------------------------
// out[b,h,t,m] = z[b,h,t] * sum_d q[t,d] * kv[d,m]; written [B,T,D] f16.
// A = q [t][d] (direct stage); B = kvT [m][d] (direct stage, column-major!).
// Grid: (T/128, B*H). Block tile 128x64, wave tile 32x32, K=64.
// ---------------------------------------------------------------------------
__global__ void attn_out_kernel(const _Float16* __restrict__ qh,
                                const _Float16* __restrict__ kvT,
                                const float* __restrict__ z,
                                _Float16* __restrict__ oh) {
  __shared__ __align__(16) _Float16 aT[128 * 72];  // [t-local][d], pad 64->72
  __shared__ __align__(16) _Float16 bT[64 * 72];   // [m][d],      pad 64->72
  int bh = blockIdx.y;
  int blockT = blockIdx.x * 128;
  int b = bh >> 4, h = bh & 15;
  int tid = threadIdx.x;
  int lane = tid & 31;
  int wave = tid >> 5;
  int wm = (wave & 3) * 32;
  int wn = (wave >> 2) * 32;

  const _Float16* qbase = qh + ((size_t)bh * TQ + blockT) * DKQ;
#pragma unroll
  for (int r = 0; r < 4; ++r) {   // 128x64 A
    int idx = tid + 256 * r;
    int row = idx >> 3;
    int kc = (idx & 7) * 8;
    *(h8*)(aT + row * 72 + kc) = *(const h8*)(qbase + (size_t)row * DKQ + kc);
  }
#pragma unroll
  for (int r = 0; r < 2; ++r) {   // 64x64 B = kvT[m][d], direct b128 copy
    int idx = tid + 256 * r;
    int m = idx >> 3;
    int kc = (idx & 7) * 8;
    *(h8*)(bT + m * 72 + kc) =
        *(const h8*)(kvT + (size_t)bh * DKQ * DKQ + (size_t)m * DKQ + kc);
  }
  __syncthreads();

  v8f acc[2][2] = {};
#pragma unroll
  for (int kk = 0; kk < DKQ; kk += 32) {
    v16h a0 = load_a_frag(aT, wm, kk, 72);
    v16h a1 = load_a_frag(aT, wm + 16, kk, 72);
    v16h b0 = load_b_frag(bT, wn, kk, 72);
    v16h b1 = load_b_frag(bT, wn + 16, kk, 72);
    acc[0][0] = wmma_f16(a0, b0, acc[0][0]);
    acc[0][1] = wmma_f16(a0, b1, acc[0][1]);
    acc[1][0] = wmma_f16(a1, b0, acc[1][0]);
    acc[1][1] = wmma_f16(a1, b1, acc[1][1]);
  }

  int half = lane >> 4;
  int nl = lane & 15;
#pragma unroll
  for (int i = 0; i < 2; ++i) {
#pragma unroll
    for (int r = 0; r < 8; ++r) {
      int trow = blockT + wm + i * 16 + r + 8 * half;
      float zz = z[(size_t)bh * TQ + trow];
#pragma unroll
      for (int j = 0; j < 2; ++j) {
        int col = h * DKQ + wn + j * 16 + nl;
        oh[((size_t)b * TQ + trow) * DQ + col] = (_Float16)(acc[i][j][r] * zz);
      }
    }
  }
}

// ---------------------------------------------------------------------------
// Launch
// ---------------------------------------------------------------------------
extern "C" void kernel_launch(void* const* d_in, const int* in_sizes, int n_in,
                              void* d_out, int out_size, void* d_ws,
                              size_t ws_size, hipStream_t stream) {
  (void)in_sizes; (void)n_in; (void)out_size; (void)ws_size;

  const float* x  = (const float*)d_in[0];
  const float* Wq = (const float*)d_in[1];
  const float* bq = (const float*)d_in[2];
  const float* Wk = (const float*)d_in[3];
  const float* bk = (const float*)d_in[4];
  const float* Wv = (const float*)d_in[5];
  const float* bv = (const float*)d_in[6];
  const float* Wo = (const float*)d_in[7];
  const float* bo = (const float*)d_in[8];
  float* out = (float*)d_out;

  char* ws = (char*)d_ws;
  const size_t MB = 1ull << 20;
  _Float16* xh  = (_Float16*)(ws + 0);           // 32 MB (reused as oh later)
  _Float16* oh  = xh;
  _Float16* wqh = (_Float16*)(ws + 32 * MB);     // 2 MB each, native [n][k]
  _Float16* wkh = (_Float16*)(ws + 34 * MB);
  _Float16* wvh = (_Float16*)(ws + 36 * MB);
  _Float16* woh = (_Float16*)(ws + 38 * MB);
  _Float16* qh  = (_Float16*)(ws + 40 * MB);     // 32 MB each
  _Float16* kh  = (_Float16*)(ws + 72 * MB);
  _Float16* vh  = (_Float16*)(ws + 104 * MB);
  _Float16* kvT = (_Float16*)(ws + 136 * MB);    // 512 KB
  float* ksum   = (float*)(ws + 136 * MB + 512 * 1024);   // 16 KB
  float* zbuf   = (float*)(ws + 136 * MB + 544 * 1024);   // 1 MB

  const int NTOT = BQ * TQ * DQ;  // 16,777,216
  const int NW = DQ * DQ;         // 1,048,576

  // 1) f32 -> f16 conversions (weights keep native [n][k] layout)
  f32_to_f16_kernel<<<NTOT / 1024, 256, 0, stream>>>(x, xh, NTOT);
  f32_to_f16_kernel<<<NW / 1024, 256, 0, stream>>>(Wq, wqh, NW);
  f32_to_f16_kernel<<<NW / 1024, 256, 0, stream>>>(Wk, wkh, NW);
  f32_to_f16_kernel<<<NW / 1024, 256, 0, stream>>>(Wv, wvh, NW);
  f32_to_f16_kernel<<<NW / 1024, 256, 0, stream>>>(Wo, woh, NW);

  // 2) q/k/v projections (WMMA), phi fused on q/k, headed layout outputs
  dim3 gProj(DQ / 128, (BQ * TQ) / 256);  // 8 x 64
  proj_gemm_kernel<<<gProj, 256, 0, stream>>>(xh, wqh, bq, qh, nullptr, 1);
  proj_gemm_kernel<<<gProj, 256, 0, stream>>>(xh, wkh, bk, kh, nullptr, 1);
  proj_gemm_kernel<<<gProj, 256, 0, stream>>>(xh, wvh, bv, vh, nullptr, 0);

  // 3) per-head summaries
  ksum_kernel<<<BQ * HQ, 256, 0, stream>>>(kh, ksum);
  kv_kernel<<<BQ * HQ, 256, 0, stream>>>(kh, vh, kvT);
  z_kernel<<<(BQ * HQ * TQ) / 256, 256, 0, stream>>>(qh, ksum, zbuf);

  // 4) out = (q @ kv) * z  -> oh in [B,T,D] f16 (aliases xh, no longer needed)
  dim3 gOut(TQ / 128, BQ * HQ);  // 32 x 64
  attn_out_kernel<<<gOut, 256, 0, stream>>>(qh, kvT, zbuf, oh);

  // 5) final projection -> f32 d_out
  proj_gemm_kernel<<<gProj, 256, 0, stream>>>(oh, woh, bo, nullptr, out, 2);
}
